// SDPASimple_6863357739228
// MI455X (gfx1250) — matrix operand order repaired and checked by reading, not compiled, not run on hardware
//
#include <hip/hip_runtime.h>
#include <math.h>

// ---------------------------------------------------------------------------
// Causal GQA flash attention for MI455X (gfx1250, wave32, WMMA).
//   B=1, S=2048, H=32, HKV=8 (n_rep=4), D=128.
// Reference semantics: cache positions 0..2047 are overwritten by fresh K/V
// and positions >= 2048 are always masked, so this is plain causal attention
// over the fresh q/k/v. f32 in/out, bf16 WMMA with f32 accumulation.
// ---------------------------------------------------------------------------

typedef __attribute__((ext_vector_type(16))) __bf16   v16bf;
typedef __attribute__((ext_vector_type(8)))  float    v8f;
typedef __attribute__((ext_vector_type(4)))  unsigned v4u;

union Frag {            // one 16x32 (A) or 32x16 (B) bf16 wave fragment
  v16bf    bf;
  v4u      u4[2];
  unsigned u[8];
};

static constexpr int S       = 2048;
static constexpr int D       = 128;
static constexpr int H       = 32;
static constexpr int HKV     = 8;
static constexpr int BLOCK_M = 128;   // q rows per workgroup (8 waves x 16)
static constexpr int BK      = 32;    // keys per inner step (= WMMA K for PV)
static constexpr int WAVES   = 8;

// scale * log2(e): softmax runs in the exp2 domain (v_exp_f32)
static constexpr float SCL2 = 0.08838834764831845f * 1.4426950408889634f;

__device__ __forceinline__ unsigned short f2bf(float x) {
  unsigned u = __float_as_uint(x);
  u += 0x7fffu + ((u >> 16) & 1u);          // round-to-nearest-even
  return (unsigned short)(u >> 16);
}

// q [s][h][d] f32 -> qb [h][s][d] bf16 ; k [s][hk][d] f32 -> kb [hk][s][d] bf16
__global__ void prep_qk_kernel(const float* __restrict__ q,
                               const float* __restrict__ k,
                               unsigned short* __restrict__ qb,
                               unsigned short* __restrict__ kb) {
  int idx  = blockIdx.x * blockDim.x + threadIdx.x;   // 0 .. S*H*D-1
  int d    = idx & (D - 1);
  int rest = idx >> 7;                                // s*H + h
  int h    = rest & (H - 1);
  int s    = rest >> 5;
  qb[(h * S + s) * D + d] = f2bf(q[idx]);
  if (idx < S * HKV * D) {
    int hk = (idx >> 7) & (HKV - 1);
    int s2 = idx >> 10;
    kb[(hk * S + s2) * D + d] = f2bf(k[idx]);
  }
}

// v [s][hk][d] f32 -> vb [hk][d][s] bf16 (LDS-tiled transpose, both sides coalesced)
__global__ void prep_v_kernel(const float* __restrict__ v,
                              unsigned short* __restrict__ vb) {
  __shared__ unsigned short t[32][33];
  const int hk = blockIdx.z;
  const int s0 = blockIdx.y * 32;
  const int d0 = blockIdx.x * 32;
  const int tx = threadIdx.x, ty = threadIdx.y;       // 32 x 8
  for (int yy = 0; yy < 32; yy += 8) {
    int s = s0 + ty + yy;
    t[ty + yy][tx] = f2bf(v[(s * HKV + hk) * D + d0 + tx]);
  }
  __syncthreads();
  for (int yy = 0; yy < 32; yy += 8) {
    int d = d0 + ty + yy;
    vb[(hk * D + d) * S + s0 + tx] = t[tx][ty + yy];
  }
}

__global__ __launch_bounds__(256)
void fa_causal_kernel(const unsigned short* __restrict__ qb,
                      const unsigned short* __restrict__ kb,
                      const unsigned short* __restrict__ vb,
                      float* __restrict__ out) {
  __shared__ __align__(16) unsigned       s_kt[BK * (D / 2)];   // [key][d/2]  8 KB
  __shared__ __align__(16) unsigned       s_vt[D * (BK / 2)];   // [d][key/2]  8 KB
  __shared__ __align__(16) unsigned short s_p[WAVES][16 * BK];  // P stage     8 KB

  const int h    = blockIdx.x >> 4;            // / (S/BLOCK_M) == 16
  const int tile = blockIdx.x & 15;
  const int kvh  = h >> 2;                     // n_rep = 4
  const int i0b  = tile * BLOCK_M;
  const int tid  = threadIdx.x;
  const int w    = tid >> 5;                   // wave32!
  const int lane = tid & 31;
  const int n    = lane & 15;                  // A row / B,C column
  const int hh   = lane >> 4;                  // half-wave selector
  const int i0   = i0b + w * 16;               // this wave's first q row
  const float NEGINF = -INFINITY;

  // --- Q resident as 4 A-fragments (16x32 bf16 each), 2x b128 loads per frag
  const unsigned* q32 = (const unsigned*)qb;
  Frag qf[4];
  {
    const int qrow = (h * S + i0 + n) * (D / 2);       // dword index
    #pragma unroll
    for (int st = 0; st < 4; ++st) {
      const unsigned* p = q32 + qrow + st * 16 + (hh ? 4 : 0);
      qf[st].u4[0] = *(const v4u*)(p);                 // K = {0..7}+8*hh
      qf[st].u4[1] = *(const v4u*)(p + 8);             // K = {16..23}+8*hh
    }
  }

  // --- online-softmax state; C-layout row for slot r is (i0 + r + 8*hh)
  float m_r[8], l_r[8], a_r[8];
  v8f   o[8];
  const v8f vzero = {0.f, 0.f, 0.f, 0.f, 0.f, 0.f, 0.f, 0.f};
  #pragma unroll
  for (int r = 0; r < 8; ++r) { m_r[r] = NEGINF; l_r[r] = 0.f; }
  #pragma unroll
  for (int t = 0; t < 8; ++t) o[t] = vzero;

  const unsigned* k32 = (const unsigned*)kb + (size_t)(kvh * S) * (D / 2);
  const unsigned* v32 = (const unsigned*)vb + (size_t)(kvh * D) * (S / 2);
  const int kend   = i0b + BLOCK_M;            // exclusive causal key bound
  const int i_last = i0 + 15;

  for (int kb0 = 0; kb0 < kend; kb0 += BK) {
    __syncthreads();                            // protect LDS reuse
    // cooperative stage: K tile (32 keys x 128 d) and V tile (128 d x 32 keys)
    #pragma unroll
    for (int r = 0; r < 8; ++r) {
      int idx = tid + 256 * r;                  // 2048 dwords each
      s_kt[idx] = k32[(kb0 + (idx >> 6)) * 64 + (idx & 63)];
    }
    #pragma unroll
    for (int r = 0; r < 8; ++r) {
      int idx = tid + 256 * r;
      s_vt[idx] = v32[(idx >> 4) * (S / 2) + (kb0 >> 1) + (idx & 15)];
    }
    if (kb0 + BK < kend)                        // global_prefetch next K tile
      __builtin_prefetch(k32 + (kb0 + BK) * 64 + (tid & 63), 0, 1);
    __syncthreads();

    if (kb0 > i_last) continue;                 // wave-uniform: EXEC stays ~0xffffffff

    // --- S = Q K^T : 2 N-subtiles x 4 K-steps = 8 WMMAs
    v8f s0 = vzero, s1 = vzero;
    #pragma unroll
    for (int st = 0; st < 4; ++st) {
      Frag kf;
      const unsigned* kr = s_kt + n * 64 + st * 16 + hh * 4;   // B: K = {0..7}+8*hh, {16..23}+8*hh
      kf.u4[0] = *(const v4u*)(kr);
      kf.u4[1] = *(const v4u*)(kr + 8);
      s0 = __builtin_amdgcn_wmma_f32_16x16x32_bf16(false, qf[st].bf, false, kf.bf,
                                                   (short)0, s0, false, false);
      const unsigned* kr1 = s_kt + (16 + n) * 64 + st * 16 + hh * 4;
      kf.u4[0] = *(const v4u*)(kr1);
      kf.u4[1] = *(const v4u*)(kr1 + 8);
      s1 = __builtin_amdgcn_wmma_f32_16x16x32_bf16(false, qf[st].bf, false, kf.bf,
                                                   (short)0, s1, false, false);
    }

    // --- masked online softmax (exp2 domain); row reductions within half-wave
    #pragma unroll
    for (int r = 0; r < 8; ++r) {
      int i = i0 + r + 8 * hh;
      float x0 = (kb0 + n      <= i) ? s0[r] * SCL2 : NEGINF;
      float x1 = (kb0 + 16 + n <= i) ? s1[r] * SCL2 : NEGINF;
      float rmax = fmaxf(x0, x1);
      rmax = fmaxf(rmax, __shfl_xor(rmax, 8, 32));
      rmax = fmaxf(rmax, __shfl_xor(rmax, 4, 32));
      rmax = fmaxf(rmax, __shfl_xor(rmax, 2, 32));
      rmax = fmaxf(rmax, __shfl_xor(rmax, 1, 32));
      float nm = fmaxf(m_r[r], rmax);
      float a, p0, p1;
      if (nm == NEGINF) { a = 1.f; p0 = 0.f; p1 = 0.f; }   // row dead so far
      else {
        a  = exp2f(m_r[r] - nm);                           // v_exp_f32
        p0 = exp2f(x0 - nm);
        p1 = exp2f(x1 - nm);
      }
      float rs = p0 + p1;
      rs += __shfl_xor(rs, 8, 32);
      rs += __shfl_xor(rs, 4, 32);
      rs += __shfl_xor(rs, 2, 32);
      rs += __shfl_xor(rs, 1, 32);
      l_r[r] = l_r[r] * a + rs;
      m_r[r] = nm;
      a_r[r] = a;
      s_p[w][(r + 8 * hh) * 32 + n]      = f2bf(p0);       // stage P row-major
      s_p[w][(r + 8 * hh) * 32 + 16 + n] = f2bf(p1);
    }
    __builtin_amdgcn_wave_barrier();   // same-wave DS ops are in-order; pin order

    // --- reload P as A-fragment (16x32 bf16)
    Frag pf;
    {
      const unsigned* pp = (const unsigned*)(&s_p[w][0]) + n * 16 + (hh ? 4 : 0);
      pf.u4[0] = *(const v4u*)(pp);
      pf.u4[1] = *(const v4u*)(pp + 8);
    }

    // --- O = diag(alpha) O + P V : 8 d-tiles, 8 WMMAs
    #pragma unroll
    for (int t = 0; t < 8; ++t) {
      #pragma unroll
      for (int r = 0; r < 8; ++r) o[t][r] *= a_r[r];
      Frag vf;
      const unsigned* vr = s_vt + (t * 16 + n) * 16 + hh * 4;  // B: K = {0..7}+8*hh, {16..23}+8*hh keys
      vf.u4[0] = *(const v4u*)(vr);
      vf.u4[1] = *(const v4u*)(vr + 8);
      o[t] = __builtin_amdgcn_wmma_f32_16x16x32_bf16(false, pf.bf, false, vf.bf,
                                                     (short)0, o[t], false, false);
    }
  }

  // --- epilogue: y = O / l ; out layout [s][h*128+d]
  #pragma unroll
  for (int r = 0; r < 8; ++r) {
    float inv = 1.0f / l_r[r];
    int   i   = i0 + r + 8 * hh;
    float* op = out + (size_t)i * (H * D) + h * D + n;
    #pragma unroll
    for (int t = 0; t < 8; ++t) op[t * 16] = o[t][r] * inv;
  }
}

extern "C" void kernel_launch(void* const* d_in, const int* in_sizes, int n_in,
                              void* d_out, int out_size, void* d_ws, size_t ws_size,
                              hipStream_t stream) {
  // setup_inputs order: input_pos, q, k, v, bsz, seqlen, mask, k_cache, v_cache
  const float* q = (const float*)d_in[1];
  const float* k = (const float*)d_in[2];
  const float* v = (const float*)d_in[3];
  float* out = (float*)d_out;

  // workspace: bf16 qb[H][S][D], kb[HKV][S][D], vb[HKV][D][S]  (~24 MB)
  unsigned short* qb = (unsigned short*)d_ws;
  unsigned short* kb = qb + (size_t)H * S * D;
  unsigned short* vb = kb + (size_t)HKV * S * D;

  prep_qk_kernel<<<(H * S * D) / 256, 256, 0, stream>>>(q, k, qb, kb);
  prep_v_kernel<<<dim3(D / 32, S / 32, HKV), dim3(32, 8), 0, stream>>>(v, vb);
  fa_causal_kernel<<<H * (S / BLOCK_M), 256, 0, stream>>>(qb, kb, vb, out);
}
